// BasicTransformerBlock_19653770346978
// MI455X (gfx1250) — matrix operand order repaired and checked by reading, not compiled, not run on hardware
//
#include <hip/hip_runtime.h>
#include <math.h>

// ---------------------------------------------------------------------------
// Types for CDNA5 WMMA (wave32, v_wmma_f32_16x16x32_bf16)
// ---------------------------------------------------------------------------
typedef __bf16 bf16_t;
typedef bf16_t v8bf  __attribute__((ext_vector_type(8)));
typedef bf16_t v16bf __attribute__((ext_vector_type(16)));
typedef float  v8f   __attribute__((ext_vector_type(8)));

// Problem dims
#define Bq    4
#define SEQ   2048
#define Dm    1024
#define CTX   77
#define CTXD  768
#define Hh    16
#define DHd   64
#define INNERD 1024
#define HID   8192
#define FFD   4096
#define MROWS (Bq*SEQ)   // 8192
#define CROWS (Bq*CTX)   // 308

__device__ __forceinline__ v8f zero_v8f() {
  v8f z;
#pragma unroll
  for (int i = 0; i < 8; ++i) z[i] = 0.0f;
  return z;
}

// Two 16-byte LDS loads -> one 16-element bf16 fragment.
__device__ __forceinline__ v16bf ld_frag16(const bf16_t* p0, const bf16_t* p1) {
  v8bf a = *(const v8bf*)p0;
  v8bf b = *(const v8bf*)p1;
  v16bf r;
#pragma unroll
  for (int i = 0; i < 8; ++i) { r[i] = a[i]; r[i + 8] = b[i]; }
  return r;
}

// ---------------------------------------------------------------------------
// CDNA5 async global->LDS DMA (ASYNCcnt-tracked), 16B per lane per issue.
// ---------------------------------------------------------------------------
typedef __attribute__((address_space(3))) char lds_char_t;

__device__ __forceinline__ uint32_t lds_off(void* p) {
  return (uint32_t)(size_t)(lds_char_t*)p;   // generic -> LDS offset
}

__device__ __forceinline__ void async_b128(uint32_t lds, const bf16_t* gaddr) {
  asm volatile("global_load_async_to_lds_b128 %0, %1, off"
               :: "v"(lds), "v"(gaddr) : "memory");
}
__device__ __forceinline__ void wait_async0() {
  asm volatile("s_wait_asynccnt 0x0" ::: "memory");
}

// ---------------------------------------------------------------------------
// LayerNorm: one block per row (D=1024), fp32 -> bf16
// ---------------------------------------------------------------------------
__global__ __launch_bounds__(256) void layernorm_bf16(
    const float* __restrict__ X, const float* __restrict__ g,
    const float* __restrict__ bta, bf16_t* __restrict__ Y, int D)
{
  __shared__ float red[256];
  const float* x = X + (size_t)blockIdx.x * D;
  bf16_t* y = Y + (size_t)blockIdx.x * D;
  const int tid = threadIdx.x;
  float s = 0.f, ss = 0.f;
  for (int i = tid; i < D; i += 256) { float v = x[i]; s += v; ss += v * v; }
  red[tid] = s; __syncthreads();
  for (int o = 128; o > 0; o >>= 1) { if (tid < o) red[tid] += red[tid + o]; __syncthreads(); }
  const float mean = red[0] * (1.0f / (float)D);
  __syncthreads();
  red[tid] = ss; __syncthreads();
  for (int o = 128; o > 0; o >>= 1) { if (tid < o) red[tid] += red[tid + o]; __syncthreads(); }
  const float var = red[0] * (1.0f / (float)D) - mean * mean;
  const float inv = rsqrtf(var + 1e-5f);
  for (int i = tid; i < D; i += 256)
    y[i] = (bf16_t)((x[i] - mean) * inv * g[i] + bta[i]);
}

// ---------------------------------------------------------------------------
// Weight prep: fp32 W[K][N] -> bf16 Wt[N][K] (K-contiguous == B-frag layout)
// Tiled 32x32 through LDS, coalesced both sides. K,N multiples of 32.
// ---------------------------------------------------------------------------
__global__ __launch_bounds__(256) void transpose_w_bf16(
    const float* __restrict__ W, bf16_t* __restrict__ Wt, int K, int N)
{
  __shared__ float t[32][33];
  const int n0 = blockIdx.x * 32, k0 = blockIdx.y * 32;
  const int tx = threadIdx.x & 31, ty = threadIdx.x >> 5;   // 32 x 8
#pragma unroll
  for (int j = 0; j < 4; ++j) {
    int kr = ty + j * 8;
    t[kr][tx] = W[(size_t)(k0 + kr) * N + n0 + tx];
  }
  __syncthreads();
#pragma unroll
  for (int j = 0; j < 4; ++j) {
    int nr = ty + j * 8;
    Wt[(size_t)(n0 + nr) * K + k0 + tx] = (bf16_t)t[tx][nr];
  }
}

// fp32 -> bf16 elementwise (context)
__global__ __launch_bounds__(256) void cvt_bf16(
    const float* __restrict__ X, bf16_t* __restrict__ Y, int total)
{
  int i = blockIdx.x * 256 + threadIdx.x;
  if (i < total) Y[i] = (bf16_t)X[i];
}

// ---------------------------------------------------------------------------
// bf16 WMMA GEMM: C = A(bf16 MxK) * Wt(bf16 NxK, pre-transposed)
//                 [+bias] [+fp32 residual]; output fp32 or bf16.
// Tile 128x128x32, double-buffered LDS filled by async global->LDS DMA.
// 256 threads = 8 waves; each wave computes 64x32 via 8 WMMAs per K-step.
// ---------------------------------------------------------------------------
#define GTM 128
#define GTN 128
#define GTK 32
#define APITCH 40   // bf16 elems per LDS row (32 + 8 pad, 16B-aligned rows)
#define ABUF (GTM * APITCH)

__global__ __launch_bounds__(256) void gemm_bf16_wmma(
    const bf16_t* __restrict__ A, int lda,     // M x K
    const bf16_t* __restrict__ Wt, int ldw,    // N x K (transposed weights)
    const float* __restrict__ bias,            // length N or nullptr
    const float* __restrict__ residual,        // M x N fp32 or nullptr
    float* __restrict__ Cf,                    // fp32 out (or nullptr)
    bf16_t* __restrict__ Cb,                   // bf16 out (or nullptr)
    int M, int N, int K)
{
  __shared__ bf16_t As[2 * ABUF];
  __shared__ bf16_t Ws[2 * ABUF];

  const int tid  = threadIdx.x;
  const int lane = tid & 31;
  const int wave = tid >> 5;
  const int lr    = lane & 15;
  const int lhalf = lane >> 4;
  const int m0 = blockIdx.y * GTM;
  const int n0 = blockIdx.x * GTN;
  const int wave_m = wave & 1;
  const int wave_n = wave >> 1;

  // per-thread DMA chunk coordinates (fixed across K-steps):
  // 512 chunks of 16B per operand tile, 2 per thread per operand.
  const int c0 = tid, c1 = tid + 256;
  const int ar0 = c0 >> 2, ac0 = (c0 & 3) << 3;   // A row / bf16 col
  const int ar1 = c1 >> 2, ac1 = (c1 & 3) << 3;
  const size_t ga0 = (size_t)((m0 + ar0 < M) ? (m0 + ar0) : (M - 1)) * lda + ac0;
  const size_t ga1 = (size_t)((m0 + ar1 < M) ? (m0 + ar1) : (M - 1)) * lda + ac1;
  const size_t gw0 = (size_t)(n0 + ar0) * ldw + ac0;
  const size_t gw1 = (size_t)(n0 + ar1) * ldw + ac1;
  const uint32_t la0 = lds_off(As + ar0 * APITCH + ac0);
  const uint32_t la1 = lds_off(As + ar1 * APITCH + ac1);
  const uint32_t lw0 = lds_off(Ws + ar0 * APITCH + ac0);
  const uint32_t lw1 = lds_off(Ws + ar1 * APITCH + ac1);

  v8f acc[4][2];
#pragma unroll
  for (int mi = 0; mi < 4; ++mi)
#pragma unroll
    for (int ni = 0; ni < 2; ++ni) acc[mi][ni] = zero_v8f();

  // prologue: DMA tile 0 into buffer 0
  async_b128(la0, A + ga0);
  async_b128(la1, A + ga1);
  async_b128(lw0, Wt + gw0);
  async_b128(lw1, Wt + gw1);

  int cur = 0;
  for (int k0 = 0; k0 < K; k0 += GTK) {
    wait_async0();          // this wave's share of tile k0 is in LDS
    __syncthreads();        // whole tile staged; prev compute done everywhere
    if (k0 + GTK < K) {     // overlap: DMA tile k0+32 while computing k0
      const uint32_t bo = (cur ^ 1) * (uint32_t)(ABUF * sizeof(bf16_t));
      const int kn = k0 + GTK;
      async_b128(la0 + bo, A + ga0 + kn);
      async_b128(la1 + bo, A + ga1 + kn);
      async_b128(lw0 + bo, Wt + gw0 + kn);
      async_b128(lw1 + bo, Wt + gw1 + kn);
    }

    const bf16_t* Ab = As + cur * ABUF;
    const bf16_t* Wb = Ws + cur * ABUF;
    v16bf afrag[4];
#pragma unroll
    for (int mi = 0; mi < 4; ++mi) {
      const bf16_t* base = Ab + (wave_m * 64 + mi * 16 + lr) * APITCH;
      const int ao = lhalf ? 8 : 0;       // A: K 0..7/16..23 vs 8..15/24..31
      afrag[mi] = ld_frag16(base + ao, base + ao + 16);
    }
    v16bf bfrag[2];
#pragma unroll
    for (int ni = 0; ni < 2; ++ni) {
      const bf16_t* base = Wb + (wave_n * 32 + ni * 16 + lr) * APITCH;
      const int bo = lhalf ? 8 : 0;       // B: K 0..7/16..23 vs 8..15/24..31
      bfrag[ni] = ld_frag16(base + bo, base + bo + 16);
    }
#pragma unroll
    for (int mi = 0; mi < 4; ++mi)
#pragma unroll
      for (int ni = 0; ni < 2; ++ni)
        acc[mi][ni] = __builtin_amdgcn_wmma_f32_16x16x32_bf16(
            false, afrag[mi], false, bfrag[ni], (short)0, acc[mi][ni],
            false, false);
    cur ^= 1;
  }

  // epilogue: bias + residual, guarded stores (fp32 or bf16)
#pragma unroll
  for (int mi = 0; mi < 4; ++mi) {
#pragma unroll
    for (int ni = 0; ni < 2; ++ni) {
      const int coln = n0 + wave_n * 32 + ni * 16 + lr;
      const float bv = bias ? bias[coln] : 0.f;
#pragma unroll
      for (int r = 0; r < 8; ++r) {
        const int rowm = m0 + wave_m * 64 + mi * 16 + r + lhalf * 8;
        if (rowm < M) {
          float v = acc[mi][ni][r] + bv;
          if (residual) v += residual[(size_t)rowm * N + coln];
          if (Cf) Cf[(size_t)rowm * N + coln] = v;
          else    Cb[(size_t)rowm * N + coln] = (bf16_t)v;
        }
      }
    }
  }
}

// ---------------------------------------------------------------------------
// Flash attention (online softmax), bf16 in / bf16 out, WMMA for QK^T & P*V.
// Block = 128 threads (4 waves) = one (b,h) x 64 q rows; wave owns 16 rows.
// Q/K staged by async DMA; V transposed into LDS ([d][k], k-contiguous).
// ---------------------------------------------------------------------------
__global__ __launch_bounds__(128) void flash_attn_wmma(
    const bf16_t* __restrict__ Qg,   // (B*seqQ) x INNER
    const bf16_t* __restrict__ Kg,   // (B*seqK) x INNER
    const bf16_t* __restrict__ Vg,   // (B*seqK) x INNER
    bf16_t* __restrict__ Og,         // (B*seqQ) x INNER
    int seqQ, int seqK)
{
  __shared__ bf16_t Qs[64 * 72];
  __shared__ bf16_t Ks[32 * 72];
  __shared__ bf16_t Vt[64 * 40];
  __shared__ bf16_t Ps[4 * 16 * 40];

  const int tid   = threadIdx.x;
  const int lane  = tid & 31;
  const int wave  = tid >> 5;
  const int lr    = lane & 15;
  const int lhalf = lane >> 4;
  const int bh = blockIdx.y;
  const int b  = bh >> 4;
  const int h  = bh & 15;
  const int q0 = blockIdx.x * 64;

  const size_t qbase = ((size_t)b * seqQ + q0) * INNERD + h * DHd;
  const size_t kbase = ((size_t)b * seqK) * INNERD + h * DHd;

  // ---- async stage Q tile (64 rows x 64 d = 512 x 16B chunks, 4/thread)
#pragma unroll
  for (int i = 0; i < 4; ++i) {
    int c = tid + i * 128;
    int row = c >> 3, cc = (c & 7) << 3;
    async_b128(lds_off(Qs + row * 72 + cc), Qg + qbase + (size_t)row * INNERD + cc);
  }

  // per-thread K-chunk DMA coords (rows recomputed per chunk for clamping)
  const int kc0 = tid, kc1 = tid + 128;            // 256 chunks per K chunk
  const int kr0 = kc0 >> 3, kcc0 = (kc0 & 7) << 3;
  const int kr1 = kc1 >> 3, kcc1 = (kc1 & 7) << 3;
  const uint32_t lk0 = lds_off(Ks + kr0 * 72 + kcc0);
  const uint32_t lk1 = lds_off(Ks + kr1 * 72 + kcc1);

  float m_[8], l_[8];
#pragma unroll
  for (int r = 0; r < 8; ++r) { m_[r] = -3.0e38f; l_[r] = 0.f; }
  v8f o_[4];
#pragma unroll
  for (int dt = 0; dt < 4; ++dt) o_[dt] = zero_v8f();

  for (int kb = 0; kb < seqK; kb += 32) {
    __syncthreads();                 // all waves done with previous chunk
    // async K rows (clamped; OOB columns masked below)
    {
      int s0r = kb + kr0; if (s0r >= seqK) s0r = seqK - 1;
      int s1r = kb + kr1; if (s1r >= seqK) s1r = seqK - 1;
      async_b128(lk0, Kg + kbase + (size_t)s0r * INNERD + kcc0);
      async_b128(lk1, Kg + kbase + (size_t)s1r * INNERD + kcc1);
    }
    // V transposed into LDS: 256 chunks of 8 bf16, 2 per thread
#pragma unroll
    for (int i = 0; i < 2; ++i) {
      int c = tid + i * 128;
      int kk = c >> 3, dc = (c & 7) << 3;
      int s = kb + kk; if (s >= seqK) s = seqK - 1;
      v8bf v = *(const v8bf*)(Vg + kbase + (size_t)s * INNERD + dc);
#pragma unroll
      for (int j = 0; j < 8; ++j) Vt[(dc + j) * 40 + kk] = v[j];
    }
    wait_async0();
    __syncthreads();

    // ---- scores: 16(q) x 32(k)
    v8f s0 = zero_v8f(), s1 = zero_v8f();
#pragma unroll
    for (int dstep = 0; dstep < 2; ++dstep) {
      const int d0 = dstep * 32;
      const bf16_t* abase = Qs + (wave * 16 + lr) * 72 + d0;
      const int ao = lhalf ? 8 : 0;
      v16bf a = ld_frag16(abase + ao, abase + ao + 16);
      {
        const bf16_t* bb = Ks + lr * 72 + d0;
        const int bo = lhalf ? 8 : 0;
        v16bf bfr = ld_frag16(bb + bo, bb + bo + 16);
        s0 = __builtin_amdgcn_wmma_f32_16x16x32_bf16(false, a, false, bfr,
                                                     (short)0, s0, false, false);
      }
      {
        const bf16_t* bb = Ks + (16 + lr) * 72 + d0;
        const int bo = lhalf ? 8 : 0;
        v16bf bfr = ld_frag16(bb + bo, bb + bo + 16);
        s1 = __builtin_amdgcn_wmma_f32_16x16x32_bf16(false, a, false, bfr,
                                                     (short)0, s1, false, false);
      }
    }

    // ---- scale + mask + online softmax
    const float scale = 0.125f;
    const bool msk0 = (kb + lr) >= seqK;
    const bool msk1 = (kb + 16 + lr) >= seqK;
    float corr[8];
#pragma unroll
    for (int r = 0; r < 8; ++r) {
      float a0 = msk0 ? -3.0e38f : s0[r] * scale;
      float a1 = msk1 ? -3.0e38f : s1[r] * scale;
      float mx = fmaxf(a0, a1);
#pragma unroll
      for (int off = 1; off < 16; off <<= 1)
        mx = fmaxf(mx, __shfl_xor(mx, off, 32));
      const float mn = fmaxf(m_[r], mx);
      corr[r] = __expf(m_[r] - mn);
      const float p0 = __expf(a0 - mn);
      const float p1 = __expf(a1 - mn);
      s0[r] = p0; s1[r] = p1;
      float sum = p0 + p1;
#pragma unroll
      for (int off = 1; off < 16; off <<= 1)
        sum += __shfl_xor(sum, off, 32);
      l_[r] = l_[r] * corr[r] + sum;
      m_[r] = mn;
    }
#pragma unroll
    for (int dt = 0; dt < 4; ++dt)
#pragma unroll
      for (int r = 0; r < 8; ++r) o_[dt][r] *= corr[r];

    // ---- C-layout -> A-layout through per-wave LDS scratch
    bf16_t* P = Ps + wave * (16 * 40);
#pragma unroll
    for (int r = 0; r < 8; ++r) {
      const int prow = r + lhalf * 8;
      P[prow * 40 + lr]      = (bf16_t)s0[r];
      P[prow * 40 + 16 + lr] = (bf16_t)s1[r];
    }
    // ---- O += P(16x32) x V(32x64)
    const bf16_t* ab = P + lr * 40;
    const int ao2 = lhalf ? 8 : 0;
    v16bf pa = ld_frag16(ab + ao2, ab + ao2 + 16);
#pragma unroll
    for (int dt = 0; dt < 4; ++dt) {
      const bf16_t* bb = Vt + (dt * 16 + lr) * 40;
      const int bo = lhalf ? 8 : 0;
      v16bf vb = ld_frag16(bb + bo, bb + bo + 16);
      o_[dt] = __builtin_amdgcn_wmma_f32_16x16x32_bf16(false, pa, false, vb,
                                                       (short)0, o_[dt],
                                                       false, false);
    }
  }

  // ---- finalize
  float inv[8];
#pragma unroll
  for (int r = 0; r < 8; ++r) inv[r] = 1.0f / l_[r];
#pragma unroll
  for (int dt = 0; dt < 4; ++dt) {
#pragma unroll
    for (int r = 0; r < 8; ++r) {
      const int qr = q0 + wave * 16 + r + lhalf * 8;
      const int d  = dt * 16 + lr;
      Og[((size_t)b * seqQ + qr) * INNERD + h * DHd + d] =
          (bf16_t)(o_[dt][r] * inv[r]);
    }
  }
}

// ---------------------------------------------------------------------------
// GeGLU: g = u * gelu_exact(gate); h fp32 rows of 8192 -> bf16 rows of 4096
// ---------------------------------------------------------------------------
__global__ __launch_bounds__(256) void geglu_kernel(
    const float* __restrict__ Hb, bf16_t* __restrict__ G, long long total)
{
  long long idx = (long long)blockIdx.x * 256 + threadIdx.x;
  if (idx >= total) return;
  long long r = idx >> 12;
  int c = (int)(idx & 4095);
  float u  = Hb[r * HID + c];
  float gt = Hb[r * HID + FFD + c];
  float ge = 0.5f * gt * (1.0f + erff(gt * 0.70710678118654752f));
  G[idx] = (bf16_t)(u * ge);
}

// ---------------------------------------------------------------------------
// Launcher
// ---------------------------------------------------------------------------
extern "C" void kernel_launch(void* const* d_in, const int* in_sizes, int n_in,
                              void* d_out, int out_size, void* d_ws, size_t ws_size,
                              hipStream_t stream) {
  (void)in_sizes; (void)n_in; (void)out_size; (void)ws_size;
  const float* x      = (const float*)d_in[0];
  const float* ctx    = (const float*)d_in[1];
  const float* ln1_g  = (const float*)d_in[2];
  const float* ln1_b  = (const float*)d_in[3];
  const float* ln2_g  = (const float*)d_in[4];
  const float* ln2_b  = (const float*)d_in[5];
  const float* ln3_g  = (const float*)d_in[6];
  const float* ln3_b  = (const float*)d_in[7];
  const float* a1_wq  = (const float*)d_in[8];
  const float* a1_wk  = (const float*)d_in[9];
  const float* a1_wv  = (const float*)d_in[10];
  const float* a1_wo  = (const float*)d_in[11];
  const float* a1_bo  = (const float*)d_in[12];
  const float* a2_wq  = (const float*)d_in[13];
  const float* a2_wk  = (const float*)d_in[14];
  const float* a2_wv  = (const float*)d_in[15];
  const float* a2_wo  = (const float*)d_in[16];
  const float* a2_bo  = (const float*)d_in[17];
  const float* ff_w1  = (const float*)d_in[18];
  const float* ff_b1  = (const float*)d_in[19];
  const float* ff_w2  = (const float*)d_in[20];
  const float* ff_b2  = (const float*)d_in[21];
  float* out = (float*)d_out;

  // ---- workspace layout: fp32 region, then bf16 region
  float* x2 = (float*)d_ws;                        // 8192x1024
  float* x3 = x2 + (size_t)MROWS * Dm;             // 8192x1024
  float* hb = x3 + (size_t)MROWS * Dm;             // 8192x8192 fp32
  bf16_t* bp   = (bf16_t*)(hb + (size_t)MROWS * HID);
  bf16_t* xnb  = bp;                               // 8192x1024
  bf16_t* qb   = xnb  + (size_t)MROWS * Dm;
  bf16_t* kb   = qb   + (size_t)MROWS * Dm;
  bf16_t* vb   = kb   + (size_t)MROWS * Dm;
  bf16_t* attb = vb   + (size_t)MROWS * Dm;
  bf16_t* gb   = attb + (size_t)MROWS * Dm;        // 8192x4096
  bf16_t* ctxb = gb   + (size_t)MROWS * FFD;       // 308x768
  bf16_t* wq1t = ctxb + (size_t)CROWS * CTXD;      // transposed bf16 weights
  bf16_t* wk1t = wq1t + (size_t)Dm * Dm;
  bf16_t* wv1t = wk1t + (size_t)Dm * Dm;
  bf16_t* wo1t = wv1t + (size_t)Dm * Dm;
  bf16_t* wq2t = wo1t + (size_t)Dm * Dm;
  bf16_t* wk2t = wq2t + (size_t)Dm * Dm;           // 1024x768
  bf16_t* wv2t = wk2t + (size_t)Dm * CTXD;
  bf16_t* wo2t = wv2t + (size_t)Dm * CTXD;
  bf16_t* w1t  = wo2t + (size_t)Dm * Dm;           // 8192x1024
  bf16_t* w2t  = w1t  + (size_t)HID * Dm;          // 1024x4096

  const dim3 blk256(256), blk128(128);
  const dim3 gemmGridD(Dm / GTN, MROWS / GTM);                 // (8, 64)
  const dim3 gemmGridCtx(Dm / GTN, (CROWS + GTM - 1) / GTM);   // (8, 3)
  const dim3 gemmGridFF1(HID / GTN, MROWS / GTM);              // (64, 64)
  const dim3 flashGrid(SEQ / 64, Bq * Hh);                     // (32, 64)

  // ---- one-time weight prep (fp32 KxN -> bf16 NxK) + context to bf16
  transpose_w_bf16<<<dim3(Dm/32,  Dm/32),  blk256, 0, stream>>>(a1_wq, wq1t, Dm, Dm);
  transpose_w_bf16<<<dim3(Dm/32,  Dm/32),  blk256, 0, stream>>>(a1_wk, wk1t, Dm, Dm);
  transpose_w_bf16<<<dim3(Dm/32,  Dm/32),  blk256, 0, stream>>>(a1_wv, wv1t, Dm, Dm);
  transpose_w_bf16<<<dim3(Dm/32,  Dm/32),  blk256, 0, stream>>>(a1_wo, wo1t, Dm, Dm);
  transpose_w_bf16<<<dim3(Dm/32,  Dm/32),  blk256, 0, stream>>>(a2_wq, wq2t, Dm, Dm);
  transpose_w_bf16<<<dim3(Dm/32,  CTXD/32),blk256, 0, stream>>>(a2_wk, wk2t, CTXD, Dm);
  transpose_w_bf16<<<dim3(Dm/32,  CTXD/32),blk256, 0, stream>>>(a2_wv, wv2t, CTXD, Dm);
  transpose_w_bf16<<<dim3(Dm/32,  Dm/32),  blk256, 0, stream>>>(a2_wo, wo2t, Dm, Dm);
  transpose_w_bf16<<<dim3(HID/32, Dm/32),  blk256, 0, stream>>>(ff_w1, w1t, Dm, HID);
  transpose_w_bf16<<<dim3(Dm/32,  FFD/32), blk256, 0, stream>>>(ff_w2, w2t, FFD, Dm);
  cvt_bf16<<<(CROWS * CTXD + 255) / 256, blk256, 0, stream>>>(ctx, ctxb, CROWS * CTXD);

  // ---- self attention
  layernorm_bf16<<<MROWS, blk256, 0, stream>>>(x, ln1_g, ln1_b, xnb, Dm);
  gemm_bf16_wmma<<<gemmGridD, blk256, 0, stream>>>(xnb, Dm, wq1t, Dm, nullptr, nullptr, nullptr, qb, MROWS, Dm, Dm);
  gemm_bf16_wmma<<<gemmGridD, blk256, 0, stream>>>(xnb, Dm, wk1t, Dm, nullptr, nullptr, nullptr, kb, MROWS, Dm, Dm);
  gemm_bf16_wmma<<<gemmGridD, blk256, 0, stream>>>(xnb, Dm, wv1t, Dm, nullptr, nullptr, nullptr, vb, MROWS, Dm, Dm);
  flash_attn_wmma<<<flashGrid, blk128, 0, stream>>>(qb, kb, vb, attb, SEQ, SEQ);
  gemm_bf16_wmma<<<gemmGridD, blk256, 0, stream>>>(attb, Dm, wo1t, Dm, a1_bo, x, x2, nullptr, MROWS, Dm, Dm);

  // ---- cross attention
  layernorm_bf16<<<MROWS, blk256, 0, stream>>>(x2, ln2_g, ln2_b, xnb, Dm);
  gemm_bf16_wmma<<<gemmGridD, blk256, 0, stream>>>(xnb, Dm, wq2t, Dm, nullptr, nullptr, nullptr, qb, MROWS, Dm, Dm);
  gemm_bf16_wmma<<<gemmGridCtx, blk256, 0, stream>>>(ctxb, CTXD, wk2t, CTXD, nullptr, nullptr, nullptr, kb, CROWS, Dm, CTXD);
  gemm_bf16_wmma<<<gemmGridCtx, blk256, 0, stream>>>(ctxb, CTXD, wv2t, CTXD, nullptr, nullptr, nullptr, vb, CROWS, Dm, CTXD);
  flash_attn_wmma<<<flashGrid, blk128, 0, stream>>>(qb, kb, vb, attb, SEQ, CTX);
  gemm_bf16_wmma<<<gemmGridD, blk256, 0, stream>>>(attb, Dm, wo2t, Dm, a2_bo, x2, x3, nullptr, MROWS, Dm, Dm);

  // ---- GeGLU feed-forward
  layernorm_bf16<<<MROWS, blk256, 0, stream>>>(x3, ln3_g, ln3_b, xnb, Dm);
  gemm_bf16_wmma<<<gemmGridFF1, blk256, 0, stream>>>(xnb, Dm, w1t, Dm, ff_b1, nullptr, hb, nullptr, MROWS, HID, Dm);
  const long long gtotal = (long long)MROWS * FFD;
  geglu_kernel<<<(unsigned)((gtotal + 255) / 256), blk256, 0, stream>>>(hb, gb, gtotal);
  gemm_bf16_wmma<<<gemmGridD, blk256, 0, stream>>>(gb, FFD, w2t, FFD, ff_b2, x3, out, nullptr, MROWS, Dm, FFD);
}
